// LlamaAttention_31026843747196
// MI455X (gfx1250) — matrix operand order repaired and checked by reading, not compiled, not run on hardware
//
#include <hip/hip_runtime.h>

typedef __bf16 bf16;
typedef __bf16 v16bf __attribute__((ext_vector_type(16)));
typedef __bf16 v8bf  __attribute__((ext_vector_type(8)));
typedef float  v8f   __attribute__((ext_vector_type(8)));

constexpr int SQ    = 2048;
constexpr int HID   = 4096;
constexpr int NHEAD = 32;
constexpr int HD    = 128;

// GEMM workgroup tile
constexpr int BM = 128, BN = 128, BK = 32;
constexpr int LROW = BK + 8;     // padded LDS row in bf16 elems (80B): conflict-free, 16B aligned
constexpr int PST  = 40;         // padded P-tile row (bf16 elems) in attention

union FragU { v16bf v; v8bf h[2]; };

__device__ __forceinline__ v8f vzero8() {
  v8f v;
#pragma unroll
  for (int i = 0; i < 8; ++i) v[i] = 0.0f;
  return v;
}

__device__ __forceinline__ v8f wmma_bf16(v16bf a, v16bf b, v8f c) {
  // D(f32 16x16) = A(bf16 16x32) * B(bf16 32x16) + C
  return __builtin_amdgcn_wmma_f32_16x16x32_bf16(false, a, false, b, (short)0, c, false, false);
}

// convert 16 staged f32 (4 x float4) to bf16 and store one 32B LDS row chunk
__device__ __forceinline__ void cvt_store16(bf16* dst, const float4& f0, const float4& f1,
                                            const float4& f2, const float4& f3) {
  v8bf lo, hv;
  lo[0]=(bf16)f0.x; lo[1]=(bf16)f0.y; lo[2]=(bf16)f0.z; lo[3]=(bf16)f0.w;
  lo[4]=(bf16)f1.x; lo[5]=(bf16)f1.y; lo[6]=(bf16)f1.z; lo[7]=(bf16)f1.w;
  hv[0]=(bf16)f2.x; hv[1]=(bf16)f2.y; hv[2]=(bf16)f2.z; hv[3]=(bf16)f2.w;
  hv[4]=(bf16)f3.x; hv[5]=(bf16)f3.y; hv[6]=(bf16)f3.z; hv[7]=(bf16)f3.w;
  *(v8bf*)dst       = lo;
  *(v8bf*)(dst + 8) = hv;
}

// CDNA5 async copy: 16B global -> LDS, tracked by ASYNCcnt (no VGPR round-trip)
__device__ __forceinline__ void async_copy_b128(bf16* dst_lds, const bf16* src_glb) {
  const unsigned lds_addr = (unsigned)(uintptr_t)dst_lds;   // low 32 bits of generic ptr = LDS offset
  asm volatile("global_load_async_to_lds_b128 %0, %1, off"
               :: "v"(lds_addr), "v"(src_glb) : "memory");
}
__device__ __forceinline__ void wait_asynccnt0() {
  asm volatile("s_wait_asynccnt 0x0" ::: "memory");
}

// ---------------------------------------------------------------------------
// C = A(M=2048 x K=4096) * W^T   with W stored [N=4096][K=4096] row-major f32.
// Double-buffered LDS pipeline; f32 sources converted to bf16 through VGPRs,
// bf16 A (context GEMM) streamed with global_load_async_to_lds_b128.
// OUTMODE: 0 = bf16 [M][HID], 1 = bf16 transposed [N][SQ] (V), 2 = f32 out.
// ---------------------------------------------------------------------------
template <bool A_IS_F32, int OUTMODE>
__global__ void __launch_bounds__(256)
gemm_xwt_kernel(const void* __restrict__ Aptr, const float* __restrict__ W,
                void* __restrict__ Outp)
{
  __shared__ __align__(16) bf16 As[2][BM * LROW];
  __shared__ __align__(16) bf16 Bs[2][BN * LROW];

  const int t    = threadIdx.x;
  const int wid  = t >> 5;          // 8 waves
  const int lane = t & 31;
  const int wm   = wid >> 2;        // 0..1  (64-row strip)
  const int wn   = wid & 3;         // 0..3  (32-col strip)
  const int lcol = lane & 15;
  const int hi   = lane >> 4;

  const int m0 = blockIdx.y * BM;
  const int n0 = blockIdx.x * BN;

  const int ldrow = t >> 1;         // 0..127: tile row this thread stages
  const int ldhal = t & 1;          // which 16-element K half

  // per-thread staging source pointers (advance by BK per K-step)
  const float* aF = (const float*)Aptr + (size_t)(m0 + ldrow) * HID + ldhal * 16;
  const bf16*  aB = (const bf16*) Aptr + (size_t)(m0 + ldrow) * HID + ldhal * 16;
  const float* wF = W + (size_t)(n0 + ldrow) * HID + ldhal * 16;

  bf16* dstA[2] = { &As[0][ldrow * LROW + ldhal * 16], &As[1][ldrow * LROW + ldhal * 16] };
  bf16* dstB[2] = { &Bs[0][ldrow * LROW + ldhal * 16], &Bs[1][ldrow * LROW + ldhal * 16] };

  v8f acc[4][2];
#pragma unroll
  for (int mi = 0; mi < 4; ++mi)
#pragma unroll
    for (int ni = 0; ni < 2; ++ni) acc[mi][ni] = vzero8();

  constexpr int NT = HID / BK;      // 128 K-steps

  // ---- prologue: stage tile 0 into buffer 0 ----
  if constexpr (A_IS_F32) {
    const float4* s4 = (const float4*)aF;
    cvt_store16(dstA[0], s4[0], s4[1], s4[2], s4[3]);
  } else {
    async_copy_b128(dstA[0], aB);
    async_copy_b128(dstA[0] + 8, aB + 8);
  }
  {
    const float4* s4 = (const float4*)wF;
    cvt_store16(dstB[0], s4[0], s4[1], s4[2], s4[3]);
  }
  if constexpr (!A_IS_F32) wait_asynccnt0();
  __syncthreads();

  for (int kt = 0; kt < NT; ++kt) {
    const int cur = kt & 1;
    const int nxt = cur ^ 1;
    const bool have_next = (kt + 1 < NT);

    // ---- issue next tile's global loads early (latency hidden by WMMAs) ----
    float4 ra0, ra1, ra2, ra3, rb0, rb1, rb2, rb3;
    if (have_next) {
      if constexpr (A_IS_F32) {
        const float4* s4 = (const float4*)(aF + (kt + 1) * BK);
        ra0 = s4[0]; ra1 = s4[1]; ra2 = s4[2]; ra3 = s4[3];
      } else {
        const bf16* s = aB + (kt + 1) * BK;
        async_copy_b128(dstA[nxt], s);
        async_copy_b128(dstA[nxt] + 8, s + 8);
      }
      const float4* t4 = (const float4*)(wF + (kt + 1) * BK);
      rb0 = t4[0]; rb1 = t4[1]; rb2 = t4[2]; rb3 = t4[3];
    }
    if (kt + 2 < NT) {               // global_prefetch_b8 two steps ahead
      if constexpr (A_IS_F32) __builtin_prefetch(aF + (kt + 2) * BK, 0, 3);
      else                    __builtin_prefetch(aB + (kt + 2) * BK, 0, 3);
      __builtin_prefetch(wF + (kt + 2) * BK, 0, 3);
    }

    // ---- WMMA on current buffer (contiguous k = 16*hi + s in both A and B fragments) ----
    FragU bfg[2];
#pragma unroll
    for (int ni = 0; ni < 2; ++ni) {
      const bf16* bp = &Bs[cur][(wn * 32 + ni * 16 + lcol) * LROW + 16 * hi];
      bfg[ni].h[0] = *(const v8bf*)bp;                  // K = 16*hi .. +7
      bfg[ni].h[1] = *(const v8bf*)(bp + 8);            // K = 16*hi+8 .. +15
    }
#pragma unroll
    for (int mi = 0; mi < 4; ++mi) {
      FragU af;
      const bf16* ap = &As[cur][(wm * 64 + mi * 16 + lcol) * LROW];
      af.h[0] = *(const v8bf*)(ap + 16 * hi);           // K = 16*hi .. +7
      af.h[1] = *(const v8bf*)(ap + 16 * hi + 8);       // K = 16*hi+8 .. +15
      acc[mi][0] = wmma_bf16(af.v, bfg[0].v, acc[mi][0]);
      acc[mi][1] = wmma_bf16(af.v, bfg[1].v, acc[mi][1]);
    }

    // ---- store staged tile into the other buffer ----
    if (have_next) {
      if constexpr (A_IS_F32) cvt_store16(dstA[nxt], ra0, ra1, ra2, ra3);
      cvt_store16(dstB[nxt], rb0, rb1, rb2, rb3);
      if constexpr (!A_IS_F32) wait_asynccnt0();
    }
    __syncthreads();
  }

  // ---- epilogue (C layout: VGPR r -> M = r + 8*hi, N = lane%16) ----
#pragma unroll
  for (int mi = 0; mi < 4; ++mi) {
#pragma unroll
    for (int ni = 0; ni < 2; ++ni) {
#pragma unroll
      for (int r = 0; r < 8; ++r) {
        const int mrow = m0 + wm * 64 + mi * 16 + r + 8 * hi;
        const int ncol = n0 + wn * 32 + ni * 16 + lcol;
        const float val = acc[mi][ni][r];
        if (OUTMODE == 0) {
          ((bf16*)Outp)[(size_t)mrow * HID + ncol] = (bf16)val;
        } else if (OUTMODE == 1) {
          ((bf16*)Outp)[(size_t)ncol * SQ + mrow] = (bf16)val;   // V transposed per head
        } else {
          ((float*)Outp)[(size_t)mrow * HID + ncol] = val;
        }
      }
    }
  }
}

// ---------------------------------------------------------------------------
// RoPE applied in place to bf16 Q and K (fp32 math).
// ---------------------------------------------------------------------------
__global__ void __launch_bounds__(256)
rope_kernel(bf16* __restrict__ Qb, bf16* __restrict__ Kb, const int* __restrict__ pos)
{
  const int idx = blockIdx.x * blockDim.x + threadIdx.x;   // S * NHEAD * 64 threads
  const int j = idx & 63;
  const int h = (idx >> 6) & (NHEAD - 1);
  const int s = idx >> 11;
  if (s >= SQ) return;

  const float p   = (float)pos[s];
  const float inv = __expf(-9.210340371976184f * (float)(2 * j) / (float)HD); // 10000^(-2j/128)
  const float ang = p * inv;
  float sn, cs;
  __sincosf(ang, &sn, &cs);

  const size_t base = (size_t)s * HID + h * HD + j;
  {
    float a = (float)Qb[base], b = (float)Qb[base + 64];
    Qb[base]      = (bf16)(a * cs - b * sn);
    Qb[base + 64] = (bf16)(b * cs + a * sn);
  }
  {
    float a = (float)Kb[base], b = (float)Kb[base + 64];
    Kb[base]      = (bf16)(a * cs - b * sn);
    Kb[base + 64] = (bf16)(b * cs + a * sn);
  }
}

// ---------------------------------------------------------------------------
// Flash attention: one wave per 16-query tile, key blocks of 32 (K=32 WMMA
// for P@V). Online softmax via wave32 xor-shuffles across 16-lane halves.
// ---------------------------------------------------------------------------
__global__ void __launch_bounds__(128)
attn_kernel(const bf16* __restrict__ Q, const bf16* __restrict__ K,
            const bf16* __restrict__ Vt, bf16* __restrict__ Ctx)
{
  __shared__ __align__(16) bf16 Pl[4][16 * PST];

  const int wid  = threadIdx.x >> 5;
  const int lane = threadIdx.x & 31;
  const int lcol = lane & 15;
  const int hi   = lane >> 4;
  const int h    = blockIdx.x;
  const int qt   = blockIdx.y * 4 + wid;
  const int q0   = qt * 16;
  const float scale = 0.08838834764831845f; // 1/sqrt(128)

  // Q fragments for full head dim: 4 K-chunks of 32
  FragU qa[4];
  const bf16* qrow = Q + (size_t)(q0 + lcol) * HID + h * HD;
#pragma unroll
  for (int kc = 0; kc < 4; ++kc) {
    qa[kc].h[0] = *(const v8bf*)(qrow + kc * 32 + 16 * hi);
    qa[kc].h[1] = *(const v8bf*)(qrow + kc * 32 + 16 * hi + 8);
  }

  v8f o[8];
#pragma unroll
  for (int d = 0; d < 8; ++d) o[d] = vzero8();
  float M[8], L[8];
#pragma unroll
  for (int r = 0; r < 8; ++r) { M[r] = -3.0e38f; L[r] = 0.0f; }

  const int nkb = q0 / 32 + 1;     // causal: key blocks 0 .. floor((q0+15)/32)
  bf16* pbuf = &Pl[wid][0];

  for (int kb = 0; kb < nkb; ++kb) {
    const int key0 = kb * 32;
    v8f sc[2];
    sc[0] = vzero8(); sc[1] = vzero8();

    // S = Q * K^T  (two 16-key tiles)
#pragma unroll
    for (int tt = 0; tt < 2; ++tt) {
      const bf16* krow = K + (size_t)(key0 + tt * 16 + lcol) * HID + h * HD;
#pragma unroll
      for (int kc = 0; kc < 4; ++kc) {
        FragU bk;
        bk.h[0] = *(const v8bf*)(krow + kc * 32 + 16 * hi);
        bk.h[1] = *(const v8bf*)(krow + kc * 32 + 16 * hi + 8);
        sc[tt] = wmma_bf16(qa[kc].v, bk.v, sc[tt]);
      }
    }

    const bool lastb = (kb == nkb - 1);
    // online softmax, per output row (M = q0 + r + 8*hi)
#pragma unroll
    for (int r = 0; r < 8; ++r) {
      const int qi = q0 + r + 8 * hi;
      float s0 = sc[0][r] * scale;
      float s1 = sc[1][r] * scale;
      if (lastb) {
        s0 = (key0 + lcol      > qi) ? -3.0e38f : s0;
        s1 = (key0 + 16 + lcol > qi) ? -3.0e38f : s1;
      }
      float mt = fmaxf(s0, s1);
      mt = fmaxf(mt, __shfl_xor(mt, 1, 32));
      mt = fmaxf(mt, __shfl_xor(mt, 2, 32));
      mt = fmaxf(mt, __shfl_xor(mt, 4, 32));
      mt = fmaxf(mt, __shfl_xor(mt, 8, 32));
      const float mnew  = fmaxf(M[r], mt);
      const float alpha = __expf(M[r] - mnew);
      const float p0 = __expf(s0 - mnew);
      const float p1 = __expf(s1 - mnew);
      float rs = p0 + p1;
      rs += __shfl_xor(rs, 1, 32);
      rs += __shfl_xor(rs, 2, 32);
      rs += __shfl_xor(rs, 4, 32);
      rs += __shfl_xor(rs, 8, 32);
      L[r] = L[r] * alpha + rs;
      M[r] = mnew;
#pragma unroll
      for (int d = 0; d < 8; ++d) o[d][r] *= alpha;
      // C-layout -> LDS (per-wave staging, in-order DS makes this safe)
      pbuf[(r + 8 * hi) * PST + lcol]      = (bf16)p0;
      pbuf[(r + 8 * hi) * PST + 16 + lcol] = (bf16)p1;
    }

    // A-layout P fragment from LDS
    FragU pa;
    pa.h[0] = *(const v8bf*)(pbuf + lcol * PST + 16 * hi);
    pa.h[1] = *(const v8bf*)(pbuf + lcol * PST + 16 * hi + 8);

    // O += P(16x32) * V(32x16) per 16-wide d-subtile; Vt is [h*HD+d][SQ]
#pragma unroll
    for (int d = 0; d < 8; ++d) {
      const bf16* vrow = Vt + (size_t)(h * HD + d * 16 + lcol) * SQ + key0 + 16 * hi;
      FragU bv;
      bv.h[0] = *(const v8bf*)vrow;
      bv.h[1] = *(const v8bf*)(vrow + 8);
      o[d] = wmma_bf16(pa.v, bv.v, o[d]);
    }
  }

  // normalize and store context [s][h*HD + d]
#pragma unroll
  for (int d = 0; d < 8; ++d) {
#pragma unroll
    for (int r = 0; r < 8; ++r) {
      const float val = o[d][r] / L[r];
      Ctx[(size_t)(q0 + r + 8 * hi) * HID + h * HD + d * 16 + lcol] = (bf16)val;
    }
  }
}

// ---------------------------------------------------------------------------
extern "C" void kernel_launch(void* const* d_in, const int* in_sizes, int n_in,
                              void* d_out, int out_size, void* d_ws, size_t ws_size,
                              hipStream_t stream)
{
  (void)in_sizes; (void)n_in; (void)out_size; (void)ws_size;

  const float* hs  = (const float*)d_in[0];
  const int*   pos = (const int*)d_in[1];
  const float* wq  = (const float*)d_in[2];
  const float* wk  = (const float*)d_in[3];
  const float* wv  = (const float*)d_in[4];
  const float* wo  = (const float*)d_in[5];

  char* ws = (char*)d_ws;
  const size_t SZ = (size_t)SQ * HID * sizeof(bf16);   // 16 MiB per tensor
  bf16* qb  = (bf16*)(ws + 0 * SZ);
  bf16* kb  = (bf16*)(ws + 1 * SZ);
  bf16* vt  = (bf16*)(ws + 2 * SZ);   // V transposed [HID][SQ]
  bf16* ctx = (bf16*)(ws + 3 * SZ);

  dim3 ggrid(HID / BN, SQ / BM);      // (32, 16)
  gemm_xwt_kernel<true, 0><<<ggrid, 256, 0, stream>>>(hs, wq, qb);
  gemm_xwt_kernel<true, 0><<<ggrid, 256, 0, stream>>>(hs, wk, kb);
  gemm_xwt_kernel<true, 1><<<ggrid, 256, 0, stream>>>(hs, wv, vt);

  rope_kernel<<<(SQ * NHEAD * 64) / 256, 256, 0, stream>>>(qb, kb, pos);

  attn_kernel<<<dim3(NHEAD, SQ / 64), 128, 0, stream>>>(qb, kb, vt, ctx);

  gemm_xwt_kernel<false, 2><<<ggrid, 256, 0, stream>>>(ctx, wo, (float*)d_out);
}
